// LightGCN_90237262889393
// MI455X (gfx1250) — compile-verified, matrix operand order checked by reading, not run on hardware
//
#include <hip/hip_runtime.h>
#include <hip/hip_bf16.h>

typedef __attribute__((ext_vector_type(16))) _Float16 v16h;
typedef __attribute__((ext_vector_type(8)))  float    v8f;

#define NUM_USERS_C 100000
#define NUM_ITEMS_C 50000
#define N_NODES_C   150000
#define E_EDGES     1250000
#define XFLOATS     9600000   // N_NODES * 64

// ---------------------------------------------------------------------------
// Edge MLP with WMMA: w = sigmoid(relu([attr|w0] @ W1 + b1) @ W2 + b2)
// One wave = 16 edges. A = 16x32 f16 (K padded 8->32), B = two 32x16 tiles of W1.
// ---------------------------------------------------------------------------
__global__ void edge_mlp_wmma(const float* __restrict__ edge_attr,
                              const float* __restrict__ ew_init,
                              const float* __restrict__ w1p,
                              const float* __restrict__ b1p,
                              const float* __restrict__ w2p,
                              const float* __restrict__ b2p,
                              float* __restrict__ w_edge) {
    const int  lane = threadIdx.x & 31;
    const int  ln   = lane & 15;          // N column within tile / A row M (lanes 0-15)
    const bool hiM  = lane >= 16;         // D rows M=8..15 live in lanes 16-31
    const long wave = (long)blockIdx.x * (blockDim.x >> 5) + (threadIdx.x >> 5);
    const long ebase = wave * 16;

    // A: lanes 0-15 hold row M=ln, halves 0..7 = K0..7 (real features), 8..15 = K16..23 (zero pad)
    //    lanes 16-31 hold K8..15 / K24..31 -> all zero pad
    v16h a = {};
    if (!hiM) {
        long e = ebase + ln;
        if (e < E_EDGES) {
            const float* ap = edge_attr + e * 7;
            #pragma unroll
            for (int k = 0; k < 7; ++k) a[k] = (_Float16)ap[k];
            a[7] = (_Float16)ew_init[e];
        }
    }
    // B tiles of W1 [K=8, N=32]: lane ln holds column, halves 0..7 = k=0..7
    v16h bt0 = {}, bt1 = {};
    if (!hiM) {
        #pragma unroll
        for (int k = 0; k < 8; ++k) {
            bt0[k] = (_Float16)w1p[k * 32 + ln];
            bt1[k] = (_Float16)w1p[k * 32 + 16 + ln];
        }
    }

    v8f c0 = {}, c1 = {};
    // (neg_a, A, neg_b, B, c_mod, C, reuse_a, reuse_b)
    c0 = __builtin_amdgcn_wmma_f32_16x16x32_f16(false, a, false, bt0, (short)0, c0, false, false);
    c1 = __builtin_amdgcn_wmma_f32_16x16x32_f16(false, a, false, bt1, (short)0, c1, false, false);

    // D layout: lane -> column N = ln; c[j] -> row M = j + (hiM ? 8 : 0).
    // Layer 2: per-edge dot over 32 hidden units = per-lane partial + 16-lane xor reduce.
    const float bias0 = b1p[ln],      bias1 = b1p[16 + ln];
    const float w2a   = w2p[ln],      w2b   = w2p[16 + ln];
    float p[8];
    #pragma unroll
    for (int j = 0; j < 8; ++j) {
        float h0 = fmaxf(c0[j] + bias0, 0.0f);
        float h1 = fmaxf(c1[j] + bias1, 0.0f);
        p[j] = h0 * w2a + h1 * w2b;
    }
    #pragma unroll
    for (int m = 1; m < 16; m <<= 1) {
        #pragma unroll
        for (int j = 0; j < 8; ++j) p[j] += __shfl_xor(p[j], m, 32);
    }
    if (ln == 0) {
        const float bb   = b2p[0];
        const int   mofs = hiM ? 8 : 0;
        #pragma unroll
        for (int j = 0; j < 8; ++j) {
            long e = ebase + mofs + j;
            if (e < E_EDGES) {
                float s = p[j] + bb;
                w_edge[e] = 1.0f / (1.0f + __expf(-s));
            }
        }
    }
}

// ---------------------------------------------------------------------------
// Node features: users = l2norm(user_emb); items = l2norm(audio + 0.5*(artist+album))
// One wave per 64-dim row (float2 per lane). Writes both x0 and acc (=d_out).
// ---------------------------------------------------------------------------
__global__ void node_embed(const float* __restrict__ user_w,
                           const float* __restrict__ artist_w,
                           const float* __restrict__ album_w,
                           const float* __restrict__ item_audio,
                           const int*   __restrict__ artist_ids,
                           const int*   __restrict__ album_ids,
                           float* __restrict__ x,
                           float* __restrict__ acc) {
    const int  lane = threadIdx.x & 31;
    const long row  = (long)blockIdx.x * (blockDim.x >> 5) + (threadIdx.x >> 5);
    if (row >= N_NODES_C) return;

    float2 v;
    if (row < NUM_USERS_C) {
        v = ((const float2*)user_w)[row * 32 + lane];
    } else {
        long i = row - NUM_USERS_C;
        int  ar = artist_ids[i], al = album_ids[i];
        float2 au = ((const float2*)item_audio)[i * 32 + lane];
        float2 aa = ((const float2*)artist_w)[(long)ar * 32 + lane];
        float2 ab = ((const float2*)album_w)[(long)al * 32 + lane];
        v.x = au.x + 0.5f * (aa.x + ab.x);
        v.y = au.y + 0.5f * (aa.y + ab.y);
    }
    float s = v.x * v.x + v.y * v.y;
    #pragma unroll
    for (int m = 1; m < 32; m <<= 1) s += __shfl_xor(s, m, 32);
    float inv = 1.0f / fmaxf(sqrtf(s), 1e-12f);
    float2 o = { v.x * inv, v.y * inv };
    ((float2*)x)[row * 32 + lane]   = o;
    ((float2*)acc)[row * 32 + lane] = o;
}

__global__ void zero_f32(float4* __restrict__ p, long n4) {
    long t = (long)blockIdx.x * blockDim.x + threadIdx.x;
    if (t < n4) p[t] = make_float4(0.f, 0.f, 0.f, 0.f);
}

__global__ void deg_accum(const float* __restrict__ w_edge,
                          const int* __restrict__ col,
                          float* __restrict__ deg) {
    long e = (long)blockIdx.x * blockDim.x + threadIdx.x;
    if (e < E_EDGES) atomicAdd(&deg[col[e]], w_edge[e]);
}

__global__ void dinv_k(const float* __restrict__ deg, float* __restrict__ dinv) {
    long n = (long)blockIdx.x * blockDim.x + threadIdx.x;
    if (n < N_NODES_C) {
        float d = deg[n];
        dinv[n] = (d > 0.0f) ? rsqrtf(fmaxf(d, 1e-30f)) : 0.0f;
    }
}

__global__ void norm_k(const float* __restrict__ w_edge,
                       const float* __restrict__ dinv,
                       const int* __restrict__ row,
                       const int* __restrict__ col,
                       float* __restrict__ nrm) {
    long e = (long)blockIdx.x * blockDim.x + threadIdx.x;
    if (e < E_EDGES) nrm[e] = dinv[row[e]] * w_edge[e] * dinv[col[e]];
}

// One wave per edge: lane carries 2 dims. Gather is L2-resident (x = 38.4MB << 192MB L2).
__global__ void propagate(const float* __restrict__ xsrc,
                          float*       __restrict__ xdst,
                          const float* __restrict__ nrm,
                          const int*   __restrict__ row,
                          const int*   __restrict__ col) {
    long t = (long)blockIdx.x * blockDim.x + threadIdx.x;
    long e = t >> 5;
    int  pr = (int)(t & 31);
    if (e >= E_EDGES) return;
    int r = row[e], c = col[e];
    float nm = nrm[e];
    float2 xv = ((const float2*)xsrc)[(long)r * 32 + pr];
    float* dst = xdst + (long)c * 64 + pr * 2;
    atomicAdd(dst,     nm * xv.x);
    atomicAdd(dst + 1, nm * xv.y);
}

__global__ void accum(float4* __restrict__ acc, const float4* __restrict__ xn, long n4) {
    long t = (long)blockIdx.x * blockDim.x + threadIdx.x;
    if (t < n4) {
        float4 a = acc[t], b = xn[t];
        a.x += b.x; a.y += b.y; a.z += b.z; a.w += b.w;
        acc[t] = a;
    }
}

// out = l2norm(acc/4), in place; also writes the trailing align_loss scalar (0).
__global__ void finalize(float* __restrict__ out) {
    if (blockIdx.x == 0 && threadIdx.x == 0) out[XFLOATS] = 0.0f;
    const int  lane = threadIdx.x & 31;
    const long row  = (long)blockIdx.x * (blockDim.x >> 5) + (threadIdx.x >> 5);
    if (row >= N_NODES_C) return;
    float2 v = ((const float2*)out)[row * 32 + lane];
    v.x *= 0.25f; v.y *= 0.25f;
    float s = v.x * v.x + v.y * v.y;
    #pragma unroll
    for (int m = 1; m < 32; m <<= 1) s += __shfl_xor(s, m, 32);
    float inv = 1.0f / fmaxf(sqrtf(s), 1e-12f);
    float2 o = { v.x * inv, v.y * inv };
    ((float2*)out)[row * 32 + lane] = o;
}

extern "C" void kernel_launch(void* const* d_in, const int* in_sizes, int n_in,
                              void* d_out, int out_size, void* d_ws, size_t ws_size,
                              hipStream_t stream) {
    (void)in_sizes; (void)n_in; (void)out_size; (void)ws_size;
    const float* user_w     = (const float*)d_in[0];
    const float* artist_w   = (const float*)d_in[1];
    const float* album_w    = (const float*)d_in[2];
    const float* item_audio = (const float*)d_in[3];
    const float* edge_attr  = (const float*)d_in[4];
    const float* ew_init    = (const float*)d_in[5];
    const float* w1         = (const float*)d_in[6];
    const float* b1         = (const float*)d_in[7];
    const float* w2         = (const float*)d_in[8];
    const float* b2         = (const float*)d_in[9];
    const int*   eidx       = (const int*)d_in[10];
    const int*   artist_ids = (const int*)d_in[11];
    const int*   album_ids  = (const int*)d_in[12];
    float* out = (float*)d_out;

    // workspace layout (floats): 22M total = 88MB
    float* ws     = (float*)d_ws;
    float* w_edge = ws;                 // E
    float* nrm    = ws + 1250000;       // E
    float* deg    = ws + 2500000;       // N
    float* dinv   = ws + 2650000;       // N
    float* xa     = ws + 2800000;       // N*64
    float* xb     = ws + 12400000;      // N*64
    const int* row = eidx;
    const int* col = eidx + E_EDGES;

    const int B = 256;

    // 1) edge MLP via WMMA: 78125 waves, 8 waves/block
    edge_mlp_wmma<<<9766, B, 0, stream>>>(edge_attr, ew_init, w1, b1, w2, b2, w_edge);

    // 2) node features -> xa and acc(=d_out)
    node_embed<<<(N_NODES_C / 8), B, 0, stream>>>(user_w, artist_w, album_w, item_audio,
                                                  artist_ids, album_ids, xa, out);

    // 3) degree, dinv, per-edge norm
    zero_f32<<<(N_NODES_C / 4 + B - 1) / B, B, 0, stream>>>((float4*)deg, N_NODES_C / 4);
    deg_accum<<<(E_EDGES + B - 1) / B, B, 0, stream>>>(w_edge, col, deg);
    dinv_k<<<(N_NODES_C + B - 1) / B, B, 0, stream>>>(deg, dinv);
    norm_k<<<(E_EDGES + B - 1) / B, B, 0, stream>>>(w_edge, dinv, row, col, nrm);

    // 4) 3 propagation layers, ping-pong xa<->xb, acc += layer output
    const long n4 = XFLOATS / 4;
    const int zeroG = (int)(n4 / B);                 // 9.6M/4/256 = 9375 exact
    const int propG = (int)(((long)E_EDGES * 32) / B); // 156250 exact

    float* src = xa; float* dst = xb;
    for (int layer = 0; layer < 3; ++layer) {
        zero_f32 <<<zeroG, B, 0, stream>>>((float4*)dst, n4);
        propagate<<<propG, B, 0, stream>>>(src, dst, nrm, row, col);
        accum    <<<zeroG, B, 0, stream>>>((float4*)out, (const float4*)dst, n4);
        float* tmp = src; src = dst; dst = tmp;
    }

    // 5) final l2norm(acc/4) in place + align_loss scalar
    finalize<<<(N_NODES_C / 8), B, 0, stream>>>(out);
}
